// MaskEstimator_35802847380095
// MI455X (gfx1250) — compile-verified, matrix operand order checked by reading, not run on hardware
//
#include <hip/hip_runtime.h>
#include <hip/hip_bf16.h>
#include <cstdint>
#include <cstddef>

// ---------------------------------------------------------------------------
// Problem constants (from reference)
// ---------------------------------------------------------------------------
#define BB    8
#define FREQ  257
#define CC    2
#define TT    800
#define DD    256
#define HH    4
#define LL    6
#define FFD   2048
#define WW    15
#define PF    7
#define DK    64
#define MROWS (BB * CC * TT)   // 12800
#define KIN   288              // FREQ padded up to multiple of 32

typedef __bf16 bf16;
typedef bf16  bfx16 __attribute__((ext_vector_type(16)));
typedef bf16  bfx8  __attribute__((ext_vector_type(8)));
typedef float fx8   __attribute__((ext_vector_type(8)));

// GEMM epilogue flags
#define F_RELU    1
#define F_SIGMOID 2

// ---------------------------------------------------------------------------
// Weight prep: fp32 (batch, K, N) -> bf16 TRANSPOSED (batch, N, Kpad),
// zero-filling K in [Kdim, Kpad). GEMM B operand is then N x K row-major,
// so both A and B tiles stage into LDS with pure b128 loads/stores.
// ---------------------------------------------------------------------------
__global__ __launch_bounds__(256) void cvt_t_kernel(
    const float* __restrict__ src, bf16* __restrict__ dst,
    int Kdim, int Ndim, int Kpad, int batch)
{
  int i = blockIdx.x * 256 + threadIdx.x;
  int per = Ndim * Kpad;
  int tot = per * batch;
  if (i >= tot) return;
  int l   = i / per;
  int rem = i - l * per;
  int n   = rem / Kpad;
  int k   = rem - n * Kpad;
  bf16 v = (bf16)0.0f;
  if (k < Kdim) v = (bf16)src[((size_t)l * Kdim + k) * Ndim + n];
  dst[i] = v;
}

// ---------------------------------------------------------------------------
// magnitude: (B,F,C,T) re/im -> bf16 (BC, T, KIN) row-major, zero-padded cols
// ---------------------------------------------------------------------------
__global__ __launch_bounds__(256) void mag_kernel(
    const float* __restrict__ xr, const float* __restrict__ xi,
    bf16* __restrict__ out)
{
  int idx = blockIdx.x * 256 + threadIdx.x;
  const int tot = MROWS * KIN;
  if (idx >= tot) return;
  int f   = idx % KIN;
  int row = idx / KIN;          // bc*T + t
  int t   = row % TT;
  int bc  = row / TT;
  int c   = bc & (CC - 1);
  int b   = bc / CC;
  bf16 val = (bf16)0.0f;
  if (f < FREQ) {
    size_t off = (((size_t)b * FREQ + f) * CC + c) * TT + t;
    float re = xr[off], im = xi[off];
    val = (bf16)sqrtf(re * re + im * im);
  }
  out[idx] = val;
}

// ---------------------------------------------------------------------------
// Tiled WMMA bf16 GEMM:  out = act( A(MxK) @ B(KxN) + bias ),
// with B supplied TRANSPOSED (BT: N x K row-major, bf16).
//   BM=128, BN=128, BK=32; 256 threads = 8 waves (4 along M x 2 along N);
//   each wave computes 2x4 16x16 WMMA tiles (32 rows x 64 cols).
//   Double-buffered LDS; next tile's global loads issued before compute.
//   GUARD_N=true adds a row guard on BT staging + column guard on stores
//   (only needed for the final N=257 GEMM).
// ---------------------------------------------------------------------------
#define BM 128
#define BN 128
#define BK 32
#define PK 40   // padded LDS K-stride (breaks bank conflicts: 80B rows)

template <bool GUARD_N>
__global__ __launch_bounds__(256) void gemm_bf16_kernel(
    const bf16* __restrict__ A,
    const bf16* __restrict__ BT,
    const float* __restrict__ bias,
    float* __restrict__ outF, bf16* __restrict__ outB, int ldo,
    int N, int K, int flags)
{
  __shared__ bf16 sA[2][BM * PK];
  __shared__ bf16 sBt[2][BN * PK];

  const int tid  = threadIdx.x;
  const int lane = tid & 31;
  const int wave = tid >> 5;
  const int wm   = wave & 3;     // 0..3 -> 32-row strip
  const int wn   = wave >> 2;    // 0..1 -> 64-col strip
  const int m0   = blockIdx.y * BM;
  const int n0   = blockIdx.x * BN;

  // staging coordinates (identical pattern for A and BT) --------------------
  const int r0 = tid >> 2;             // rows 0..63   (chunk 0)
  const int r1 = r0 + 64;              // rows 64..127 (chunk 1)
  const int kc = (tid & 3) * 8;        // 0,8,16,24

  bfx8 ra0, ra1, rb0, rb1;
  bfx8 zv;
  #pragma unroll
  for (int e = 0; e < 8; ++e) zv[e] = (bf16)0.0f;

  auto fetch = [&](int k0) {
    ra0 = *(const bfx8*)(A + (size_t)(m0 + r0) * K + k0 + kc);
    ra1 = *(const bfx8*)(A + (size_t)(m0 + r1) * K + k0 + kc);
    if (!GUARD_N) {
      rb0 = *(const bfx8*)(BT + (size_t)(n0 + r0) * K + k0 + kc);
      rb1 = *(const bfx8*)(BT + (size_t)(n0 + r1) * K + k0 + kc);
    } else {
      rb0 = (n0 + r0 < N) ? *(const bfx8*)(BT + (size_t)(n0 + r0) * K + k0 + kc) : zv;
      rb1 = (n0 + r1 < N) ? *(const bfx8*)(BT + (size_t)(n0 + r1) * K + k0 + kc) : zv;
    }
  };
  auto stage = [&](int buf) {
    *(bfx8*)(sA[buf] + r0 * PK + kc) = ra0;
    *(bfx8*)(sA[buf] + r1 * PK + kc) = ra1;
    *(bfx8*)(sBt[buf] + r0 * PK + kc) = rb0;
    *(bfx8*)(sBt[buf] + r1 * PK + kc) = rb1;
  };

  fx8 acc[2][4] = {};

  const int half = lane >> 4;    // K-half selector per WMMA layout
  const int rr   = lane & 15;
  const int klo  = half * 8;

  const int nk = K / BK;
  fetch(0);
  stage(0);
  __syncthreads();

  for (int kt = 0; kt < nk; ++kt) {
    const int cur = kt & 1;
    const bool hasNext = (kt + 1 < nk);
    if (hasNext) fetch((kt + 1) * BK);   // loads in flight during WMMA work

    bfx16 afr[2], bfr[4];
    #pragma unroll
    for (int i = 0; i < 2; ++i) {
      int row = wm * 32 + i * 16 + rr;
      bfx8 lo = *(const bfx8*)(sA[cur] + row * PK + klo);
      bfx8 hi = *(const bfx8*)(sA[cur] + row * PK + 16 + klo);
      #pragma unroll
      for (int e = 0; e < 8; ++e) { afr[i][e] = lo[e]; afr[i][8 + e] = hi[e]; }
    }
    #pragma unroll
    for (int j = 0; j < 4; ++j) {
      int col = wn * 64 + j * 16 + rr;
      bfx8 lo = *(const bfx8*)(sBt[cur] + col * PK + klo);
      bfx8 hi = *(const bfx8*)(sBt[cur] + col * PK + 16 + klo);
      #pragma unroll
      for (int e = 0; e < 8; ++e) { bfr[j][e] = lo[e]; bfr[j][8 + e] = hi[e]; }
    }
    #pragma unroll
    for (int i = 0; i < 2; ++i)
      #pragma unroll
      for (int j = 0; j < 4; ++j)
        acc[i][j] = __builtin_amdgcn_wmma_f32_16x16x32_bf16(
            false, afr[i], false, bfr[j], (short)0, acc[i][j], false, false);

    if (hasNext) {
      __syncthreads();
      stage(cur ^ 1);
      __syncthreads();
    }
  }

  // --- epilogue: bias + activation, f32 and/or bf16 store ----------------
  #pragma unroll
  for (int i = 0; i < 2; ++i) {
    #pragma unroll
    for (int j = 0; j < 4; ++j) {
      int gc = n0 + wn * 64 + j * 16 + rr;
      if (GUARD_N && gc >= N) continue;
      float bv = bias[gc];
      #pragma unroll
      for (int vr = 0; vr < 8; ++vr) {
        int gr  = m0 + wm * 32 + i * 16 + half * 8 + vr;
        float v = acc[i][j][vr] + bv;
        if (flags & F_RELU)    v = fmaxf(v, 0.0f);
        if (flags & F_SIGMOID) v = 1.0f / (1.0f + __expf(-v));
        size_t o = (size_t)gr * ldo + gc;
        if (outF) outF[o] = v;
        if (outB) outB[o] = (bf16)v;
      }
    }
  }
}

// ---------------------------------------------------------------------------
// Windowed attention (W=15, DK=64): one thread per (bc, h, t). Tiny FLOP
// share -> fp32 VALU; output written bf16 (feeds Wo GEMM).
// ---------------------------------------------------------------------------
__global__ __launch_bounds__(256) void attn_kernel(
    const float* __restrict__ q, const float* __restrict__ k,
    const float* __restrict__ v, const int* __restrict__ ilens,
    bf16* __restrict__ oB)
{
  int idx = blockIdx.x * 256 + threadIdx.x;
  const int tot = BB * CC * HH * TT;
  if (idx >= tot) return;
  int t  = idx % TT;
  int h  = (idx / TT) & (HH - 1);
  int bc = idx / (TT * HH);
  int len = ilens[bc / CC];

  const float* qp = q + ((size_t)bc * TT + t) * DD + h * DK;
  float qv[DK];
  #pragma unroll
  for (int d = 0; d < DK; ++d) qv[d] = qp[d];

  float sc[WW];
  float mx = -3.4e38f;
  #pragma unroll
  for (int w = 0; w < WW; ++w) {
    int tk = t - PF + w;
    float s = -3.4e38f;
    if (tk >= 0 && tk < TT && tk < len) {
      const float* kp = k + ((size_t)bc * TT + tk) * DD + h * DK;
      float dot = 0.0f;
      #pragma unroll
      for (int d = 0; d < DK; ++d) dot += qv[d] * kp[d];
      s = dot * 0.125f;     // 1/sqrt(64)
    }
    sc[w] = s;
    mx = fmaxf(mx, s);
  }

  bf16* op = oB + ((size_t)bc * TT + t) * DD + h * DK;
  if (mx <= -3.0e38f) {     // fully masked row -> zeros (matches reference)
    #pragma unroll
    for (int d = 0; d < DK; ++d) op[d] = (bf16)0.0f;
    return;
  }

  float acc[DK] = {};
  float sum = 0.0f;
  for (int w = 0; w < WW; ++w) {
    if (sc[w] > -3.0e38f) {
      float e = __expf(sc[w] - mx);
      sum += e;
      int tk = t - PF + w;
      const float* vp = v + ((size_t)bc * TT + tk) * DD + h * DK;
      #pragma unroll
      for (int d = 0; d < DK; ++d) acc[d] += e * vp[d];
    }
  }
  float inv = 1.0f / sum;
  #pragma unroll
  for (int d = 0; d < DK; ++d) op[d] = (bf16)(acc[d] * inv);
}

// ---------------------------------------------------------------------------
// x = LayerNorm(resid + y) * g + b ; write fp32 truth and bf16 GEMM operand.
// One wave per row of 256; lane covers 8 strided elements; shfl reductions.
// ---------------------------------------------------------------------------
__global__ __launch_bounds__(256) void add_ln_kernel(
    const float* __restrict__ resid, const float* __restrict__ y,
    const float* __restrict__ g, const float* __restrict__ b,
    float* __restrict__ xF, bf16* __restrict__ xB)
{
  int wave = threadIdx.x >> 5;
  int lane = threadIdx.x & 31;
  int row  = blockIdx.x * 8 + wave;
  if (row >= MROWS) return;
  const float* rp = resid + (size_t)row * DD;
  const float* yp = y + (size_t)row * DD;

  float v[8];
  float s = 0.0f;
  #pragma unroll
  for (int i = 0; i < 8; ++i) {
    int c = lane + 32 * i;
    v[i] = rp[c] + yp[c];
    s += v[i];
  }
  #pragma unroll
  for (int off = 16; off > 0; off >>= 1) s += __shfl_xor(s, off, 32);
  float mean = s * (1.0f / DD);

  float var = 0.0f;
  #pragma unroll
  for (int i = 0; i < 8; ++i) {
    float d = v[i] - mean;
    var += d * d;
  }
  #pragma unroll
  for (int off = 16; off > 0; off >>= 1) var += __shfl_xor(var, off, 32);
  float inv = rsqrtf(var * (1.0f / DD) + 1e-12f);

  #pragma unroll
  for (int i = 0; i < 8; ++i) {
    int c = lane + 32 * i;
    float o = (v[i] - mean) * inv * g[c] + b[c];
    xF[(size_t)row * DD + c] = o;
    xB[(size_t)row * DD + c] = (bf16)o;
  }
}

// ---------------------------------------------------------------------------
// final transpose: maskF (BC,T,FREQ) fp32 -> out (B,F,C,T) fp32
// ---------------------------------------------------------------------------
__global__ __launch_bounds__(256) void out_transpose_kernel(
    const float* __restrict__ maskF, float* __restrict__ out)
{
  int idx = blockIdx.x * 256 + threadIdx.x;
  const int tot = BB * FREQ * CC * TT;
  if (idx >= tot) return;
  int t = idx % TT;
  int c = (idx / TT) % CC;
  int f = (idx / (TT * CC)) % FREQ;
  int b = idx / (TT * CC * FREQ);
  out[idx] = maskF[(((size_t)b * CC + c) * TT + t) * FREQ + f];
}

// ---------------------------------------------------------------------------
// Host orchestration
// ---------------------------------------------------------------------------
static inline size_t align256(size_t x) { return (x + 255) & ~(size_t)255; }

extern "C" void kernel_launch(void* const* d_in, const int* in_sizes, int n_in,
                              void* d_out, int out_size, void* d_ws, size_t ws_size,
                              hipStream_t stream) {
  (void)in_sizes; (void)n_in; (void)out_size; (void)ws_size;

  const float* xs_real = (const float*)d_in[0];
  const float* xs_imag = (const float*)d_in[1];
  const int*   ilens   = (const int*)  d_in[2];
  const float* W_in    = (const float*)d_in[3];
  const float* b_in    = (const float*)d_in[4];
  const float* Wq      = (const float*)d_in[5];
  const float* bq      = (const float*)d_in[6];
  const float* Wk      = (const float*)d_in[7];
  const float* bk      = (const float*)d_in[8];
  const float* Wv      = (const float*)d_in[9];
  const float* bv      = (const float*)d_in[10];
  const float* Wo      = (const float*)d_in[11];
  const float* bo      = (const float*)d_in[12];
  const float* W1      = (const float*)d_in[13];
  const float* b1      = (const float*)d_in[14];
  const float* W2      = (const float*)d_in[15];
  const float* b2      = (const float*)d_in[16];
  const float* n1_s    = (const float*)d_in[17];
  const float* n1_b    = (const float*)d_in[18];
  const float* n2_s    = (const float*)d_in[19];
  const float* n2_b    = (const float*)d_in[20];
  const float* Wm      = (const float*)d_in[21];
  const float* bm      = (const float*)d_in[22];
  float* out = (float*)d_out;

  // --- transposed bf16 weight buffer layout (elements, N x Kpad per layer) ---
  const size_t eWin = (size_t)DD * KIN;          // 256 x 288 (K padded)
  const size_t eP   = (size_t)DD * DD;           // 256 x 256
  const size_t eF1  = (size_t)FFD * DD;          // 2048 x 256 (W1^T)
  const size_t eF2  = (size_t)DD * FFD;          // 256 x 2048 (W2^T)
  const size_t eWm  = (size_t)FREQ * DD;         // 257 x 256 (Wm^T)
  const size_t oWin = 0;
  const size_t oWq  = oWin + eWin;
  const size_t oWk  = oWq + LL * eP;
  const size_t oWv  = oWk + LL * eP;
  const size_t oWo  = oWv + LL * eP;
  const size_t oW1  = oWo + LL * eP;
  const size_t oW2  = oW1 + LL * eF1;
  const size_t oWm  = oW2 + LL * eF2;
  const size_t wbfElems = oWm + eWm;

  // --- workspace layout ---
  char* base = (char*)d_ws;
  size_t cur = 0;
  auto take = [&](size_t bytes) { size_t o = cur; cur = align256(cur + bytes); return o; };
  bf16*  wbf   = (bf16*) (base + take(wbfElems * sizeof(bf16)));
  bf16*  magBF = (bf16*) (base + take((size_t)MROWS * KIN * sizeof(bf16)));
  bf16*  xBF   = (bf16*) (base + take((size_t)MROWS * DD * sizeof(bf16)));
  bf16*  oBF   = (bf16*) (base + take((size_t)MROWS * DD * sizeof(bf16)));
  bf16*  hBF   = (bf16*) (base + take((size_t)MROWS * FFD * sizeof(bf16)));
  float* xF    = (float*)(base + take((size_t)MROWS * DD * sizeof(float)));
  float* qF    = (float*)(base + take((size_t)MROWS * DD * sizeof(float)));
  float* kF    = (float*)(base + take((size_t)MROWS * DD * sizeof(float)));
  float* vF    = (float*)(base + take((size_t)MROWS * DD * sizeof(float)));
  float* yF    = (float*)(base + take((size_t)MROWS * DD * sizeof(float)));
  float* maskF = (float*)hBF;   // alias: hBF dead by the time mask is computed

  // convert + transpose weights: src (batch, K, N) fp32 -> dst (batch, N, Kpad) bf16
  auto cvtT = [&](const float* s, bf16* d, int Kdim, int Ndim, int Kpad, int batch) {
    int tot = Ndim * Kpad * batch;
    cvt_t_kernel<<<(tot + 255) / 256, 256, 0, stream>>>(s, d, Kdim, Ndim, Kpad, batch);
  };
  cvtT(W_in, wbf + oWin, FREQ, DD, KIN, 1);      // K 257 -> 288 padded
  cvtT(Wq,   wbf + oWq,  DD, DD, DD, LL);
  cvtT(Wk,   wbf + oWk,  DD, DD, DD, LL);
  cvtT(Wv,   wbf + oWv,  DD, DD, DD, LL);
  cvtT(Wo,   wbf + oWo,  DD, DD, DD, LL);
  cvtT(W1,   wbf + oW1,  DD, FFD, DD, LL);
  cvtT(W2,   wbf + oW2,  FFD, DD, FFD, LL);
  cvtT(Wm,   wbf + oWm,  DD, FREQ, DD, 1);

  // --- magnitude + transpose into padded bf16 matrix ---
  {
    int tot = MROWS * KIN;
    mag_kernel<<<tot / 256, 256, 0, stream>>>(xs_real, xs_imag, magBF);
  }

  auto gemm = [&](const bf16* A, const bf16* BT, const float* bias,
                  float* oF, bf16* oBp, int ldo, int N, int K, int flags) {
    dim3 grid((N + BN - 1) / BN, MROWS / BM);
    if (N % BN == 0) {
      gemm_bf16_kernel<false><<<grid, 256, 0, stream>>>(A, BT, bias, oF, oBp,
                                                        ldo, N, K, flags);
    } else {
      gemm_bf16_kernel<true><<<grid, 256, 0, stream>>>(A, BT, bias, oF, oBp,
                                                       ldo, N, K, flags);
    }
  };

  // input projection: x = mag @ W_in + b_in  (write fp32 truth + bf16 operand)
  gemm(magBF, wbf + oWin, b_in, xF, xBF, DD, DD, KIN, 0);

  for (int l = 0; l < LL; ++l) {
    // Q, K, V projections (fp32 outputs for the attention kernel)
    gemm(xBF, wbf + oWq + l * eP, bq + l * DD, qF, nullptr, DD, DD, DD, 0);
    gemm(xBF, wbf + oWk + l * eP, bk + l * DD, kF, nullptr, DD, DD, DD, 0);
    gemm(xBF, wbf + oWv + l * eP, bv + l * DD, vF, nullptr, DD, DD, DD, 0);
    // windowed attention -> bf16 operand for Wo
    {
      int tot = BB * CC * HH * TT;
      attn_kernel<<<(tot + 255) / 256, 256, 0, stream>>>(qF, kF, vF, ilens, oBF);
    }
    // output projection
    gemm(oBF, wbf + oWo + l * eP, bo + l * DD, yF, nullptr, DD, DD, DD, 0);
    // x = LN(x + y) [n1]
    add_ln_kernel<<<MROWS / 8, 256, 0, stream>>>(xF, yF, n1_s + l * DD, n1_b + l * DD, xF, xBF);
    // FFN: h = relu(x @ W1 + b1) (bf16 only), y = h @ W2 + b2
    gemm(xBF, wbf + oW1 + l * eF1, b1 + l * FFD, nullptr, hBF, FFD, FFD, DD, F_RELU);
    gemm(hBF, wbf + oW2 + l * eF2, b2 + l * DD, yF, nullptr, DD, DD, FFD, 0);
    // x = LN(x + ff) [n2]
    add_ln_kernel<<<MROWS / 8, 256, 0, stream>>>(xF, yF, n2_s + l * DD, n2_b + l * DD, xF, xBF);
  }

  // final mask: sigmoid(x @ Wm + bm), N=257 (guarded), then transpose out
  gemm(xBF, wbf + oWm, bm, maskF, nullptr, FREQ, FREQ, DD, F_SIGMOID);
  {
    int tot = BB * FREQ * CC * TT;
    out_transpose_kernel<<<(tot + 255) / 256, 256, 0, stream>>>(maskF, out);
  }
}
